// PerformanceLensHybrid_25615184953904
// MI455X (gfx1250) — compile-verified, matrix operand order checked by reading, not run on hardware
//
#include <hip/hip_runtime.h>
#include <hip/hip_bf16.h>
#include <math.h>

#define NNODES 65536
#define NEDGES 131072
#define NEP    (NEDGES + NNODES)   // edges + self loops
#define BGR    32
#define EPER   4096
#define HDIM   128
#define OUTD   64
#define CSTR   200                 // padded LDS row stride (f16 elems) for GRU c-matrix

typedef _Float16 v16h __attribute__((ext_vector_type(16)));
typedef _Float16 v8h  __attribute__((ext_vector_type(8)));
typedef float    v8f  __attribute__((ext_vector_type(8)));

// ---- WMMA fragment load: 16-bit A/B 16x32 layout.
// Caller passes p = row_base + kstep*32 + half*8   (half = lane>=16)
// lane<16 : K = {0..7} U {16..23};  lane>=16 : K = {8..15} U {24..31}
__device__ __forceinline__ v16h ldfrag(const _Float16* p) {
  v8h lo = *(const v8h*)(p);
  v8h hi = *(const v8h*)(p + 16);
  v16h r;
#pragma unroll
  for (int i = 0; i < 8; ++i) { r[i] = lo[i]; r[i + 8] = hi[i]; }
  return r;
}

__device__ __forceinline__ v8f wmma_f16(v16h a, v16h b, v8f c) {
  return __builtin_amdgcn_wmma_f32_16x16x32_f16(false, a, false, b, (short)0, c, false, false);
}

__device__ __forceinline__ float gelu_(float v) {
  return 0.5f * v * (1.0f + erff(v * 0.7071067811865476f));
}

// order-preserving float<->uint for atomicMax
__device__ __forceinline__ unsigned fenc(float f) {
  unsigned u = __float_as_uint(f);
  return (u & 0x80000000u) ? ~u : (u | 0x80000000u);
}
__device__ __forceinline__ float fdec(unsigned u) {
  return (u & 0x80000000u) ? __uint_as_float(u & 0x7fffffffu) : __uint_as_float(~u);
}

// ---------------- small prep kernels ----------------

// W [K][N] f32 -> Bt [N][K] f16
__global__ void k_conv_bt(const float* __restrict__ W, _Float16* __restrict__ Bt, int K, int N) {
  int i = blockIdx.x * 256 + threadIdx.x;
  if (i >= K * N) return;
  int k = i / N, n = i - k * N;
  Bt[(size_t)n * K + k] = (_Float16)W[i];
}

// h0 = concat(x, emb[node_index]) as f16 [NN][64]
__global__ void k_build_h0(const float* __restrict__ x, const int* __restrict__ nidx,
                           const float* __restrict__ emb, _Float16* __restrict__ h0) {
  int i = blockIdx.x * 256 + threadIdx.x;
  int n = i >> 6, c = i & 63;
  float v = (c < 32) ? x[(size_t)n * 32 + c] : emb[(size_t)nidx[n] * 32 + (c - 32)];
  h0[i] = (_Float16)v;
}

__global__ void k_zero_f32(float* p, int n) {
  int i = blockIdx.x * 256 + threadIdx.x;
  if (i < n) p[i] = 0.0f;
}
__global__ void k_init_ms(unsigned* m, float* s) {
  int i = blockIdx.x * 256 + threadIdx.x;
  if (i < NNODES) { m[i] = 0u; s[i] = 0.0f; }
}

// ---------------- node GEMM: h = A(f16, MxK) @ Bt^T  -> C f32 [M][128] ----------------
__global__ __launch_bounds__(256) void k_gemm_nodes(const _Float16* __restrict__ A,
                                                    const _Float16* __restrict__ Bt,
                                                    float* __restrict__ C, int K) {
  const int tid = threadIdx.x;
  const int w = tid >> 5, l = tid & 31, lj = l & 15, half = l >> 4;
  const int m0 = blockIdx.x * 128 + w * 16;
  const _Float16* arow = A + (size_t)(m0 + lj) * K;
  for (int nt = 0; nt < 8; ++nt) {
    const _Float16* brow = Bt + (size_t)(nt * 16 + lj) * K;
    v8f acc = {};
    for (int kk = 0; kk < K; kk += 32) {
      v16h af = ldfrag(arow + kk + half * 8);
      v16h bf = ldfrag(brow + kk + half * 8);
      acc = wmma_f16(af, bf, acc);
    }
    const int col = nt * 16 + lj;
#pragma unroll
    for (int v = 0; v < 8; ++v) {
      int row = m0 + v + 8 * half;
      C[(size_t)row * HDIM + col] = acc[v];
    }
  }
}

// per-node attention scores: ss = h . a_s, sd = h . a_d
__global__ void k_scores(const float* __restrict__ h, const float* __restrict__ as,
                         const float* __restrict__ ad, float* __restrict__ ss,
                         float* __restrict__ sd) {
  int i = blockIdx.x * 256 + threadIdx.x;
  const float* r = h + (size_t)i * HDIM;
  float a = 0.f, b = 0.f;
#pragma unroll 4
  for (int c = 0; c < HDIM; ++c) { float v = r[c]; a += v * as[c]; b += v * ad[c]; }
  ss[i] = a; sd[i] = b;
}

__device__ __forceinline__ void edge_sd(int e, const int* src, const int* dst, int& s, int& d) {
  if (e < NEDGES) { s = src[e]; d = dst[e]; } else { s = d = e - NEDGES; }
}

__global__ void k_edge_max(const float* __restrict__ ss, const float* __restrict__ sd,
                           const int* __restrict__ src, const int* __restrict__ dst,
                           unsigned* __restrict__ m) {
  int e = blockIdx.x * 256 + threadIdx.x;
  int s, d; edge_sd(e, src, dst, s, d);
  float v = ss[s] + sd[d];
  v = v > 0.f ? v : 0.2f * v;
  atomicMax(&m[d], fenc(v));
}

__global__ void k_edge_expsum(const float* __restrict__ ss, const float* __restrict__ sd,
                              const int* __restrict__ src, const int* __restrict__ dst,
                              const unsigned* __restrict__ m, float* __restrict__ exb,
                              float* __restrict__ ssum) {
  int e = blockIdx.x * 256 + threadIdx.x;
  int s, d; edge_sd(e, src, dst, s, d);
  float v = ss[s] + sd[d];
  v = v > 0.f ? v : 0.2f * v;
  float ex = expf(v - fdec(m[d]));
  exb[e] = ex;
  atomicAdd(&ssum[d], ex);
}

// out[dst] += h[src] * alpha  (128 lanes per edge, 2 edges per block)
__global__ void k_aggregate(const float* __restrict__ h, const int* __restrict__ src,
                            const int* __restrict__ dst, const float* __restrict__ exb,
                            const float* __restrict__ ssum, float* __restrict__ out) {
  int e = blockIdx.x * 2 + (threadIdx.x >> 7);
  int col = threadIdx.x & 127;
  int s, d; edge_sd(e, src, dst, s, d);
  float alpha = exb[e] / ssum[d];
  atomicAdd(&out[(size_t)d * HDIM + col], h[(size_t)s * HDIM + col] * alpha);
}

// hf16 = f16(gelu(out + b))
__global__ void k_finalize(const float* __restrict__ out, const float* __restrict__ b,
                           _Float16* __restrict__ hf16) {
  int i = blockIdx.x * 256 + threadIdx.x;
  int col = i & (HDIM - 1);
  hf16[i] = (_Float16)gelu_(out[i] + b[col]);
}

// ---------------- edge FC: ef = gelu([h[src],h[dst]] @ fcW + fcb) -> gru[t][b][:] f16 ----------------
__global__ __launch_bounds__(256) void k_fc_gemm(const _Float16* __restrict__ h3,
                                                 const _Float16* __restrict__ Bt,   // [128][256]
                                                 const int* __restrict__ src, const int* __restrict__ dst,
                                                 const float* __restrict__ fcb,
                                                 _Float16* __restrict__ gru) {
  const int tid = threadIdx.x;
  const int w = tid >> 5, l = tid & 31, lj = l & 15, half = l >> 4;
  const int m0 = blockIdx.x * 128 + w * 16;
  const int e = m0 + lj;
  const _Float16* rowS = h3 + (size_t)src[e] * HDIM;
  const _Float16* rowD = h3 + (size_t)dst[e] * HDIM;
  for (int nt = 0; nt < 8; ++nt) {
    const _Float16* brow = Bt + (size_t)(nt * 16 + lj) * 256;
    v8f acc = {};
#pragma unroll
    for (int kk = 0; kk < 4; ++kk) {
      v16h af = ldfrag(rowS + kk * 32 + half * 8);
      v16h bf = ldfrag(brow + kk * 32 + half * 8);
      acc = wmma_f16(af, bf, acc);
    }
#pragma unroll
    for (int kk = 0; kk < 4; ++kk) {
      v16h af = ldfrag(rowD + kk * 32 + half * 8);
      v16h bf = ldfrag(brow + 128 + kk * 32 + half * 8);
      acc = wmma_f16(af, bf, acc);
    }
    const int col = nt * 16 + lj;
    const float bias = fcb[col];
#pragma unroll
    for (int v = 0; v < 8; ++v) {
      int erow = m0 + v + 8 * half;
      int t = erow & (EPER - 1), b = erow >> 12;
      gru[((size_t)t * BGR + b) * HDIM + col] = (_Float16)gelu_(acc[v] + bias);
    }
  }
}

// ---------------- GRU scan: 1 workgroup, 8 waves, each wave owns one 16x16 tile of [32][64] ----------------
__global__ __launch_bounds__(256, 1) void k_gru(const _Float16* __restrict__ gin,   // [T][32][128] f16
                                                const _Float16* __restrict__ gwt,   // [3][64][192] f16 (N-major)
                                                const float* __restrict__ bz, const float* __restrict__ br,
                                                const float* __restrict__ bh, float* __restrict__ out) {
  __shared__ _Float16 cA[32 * CSTR];   // c = [xt(128) | hid/r*hid(64)]
  __shared__ float hid[BGR * OUTD];
  const int tid = threadIdx.x;
  const int w = tid >> 5, l = tid & 31, lj = l & 15, half = l >> 4;
  const int mt = w >> 2, nt = w & 3;
  const int N = nt * 16 + lj;                 // this lane's output column
  const float bzv = bz[N], brv = br[N], bhv = bh[N];
  const _Float16* Wz = gwt;
  const _Float16* Wr = gwt + 64 * 192;
  const _Float16* Wh = gwt + 2 * 64 * 192;
  const _Float16* zrow = Wz + (size_t)N * 192;
  const _Float16* rrow = Wr + (size_t)N * 192;
  const _Float16* hrow = Wh + (size_t)N * 192;
  const _Float16* arow = &cA[(mt * 16 + lj) * CSTR];

  for (int i = tid; i < 32 * CSTR; i += 256) cA[i] = (_Float16)0.f;
  for (int i = tid; i < BGR * OUTD; i += 256) hid[i] = 0.f;
  __syncthreads();

  for (int t = 0; t < EPER; ++t) {
    { // stream xt into cA cols [0,128)
      int idx = tid * 16;
      int b = idx >> 7, c = idx & 127;
      *(v16h*)&cA[b * CSTR + c] = *(const v16h*)(gin + (size_t)t * BGR * HDIM + idx);
    }
    __syncthreads();   // B1: xt + prev hid writes visible

    v8f zacc = {}, racc = {};
#pragma unroll
    for (int ks = 0; ks < 6; ++ks) {
      v16h af = ldfrag(arow + ks * 32 + half * 8);
      zacc = wmma_f16(af, ldfrag(zrow + ks * 32 + half * 8), zacc);
      racc = wmma_f16(af, ldfrag(rrow + ks * 32 + half * 8), racc);
    }
    __syncthreads();   // B2: all gemm reads of cA done

    float zv[8];
#pragma unroll
    for (int v = 0; v < 8; ++v) {
      int M = mt * 16 + v + 8 * half;
      zv[v] = 1.f / (1.f + expf(-(zacc[v] + bzv)));
      float r = 1.f / (1.f + expf(-(racc[v] + brv)));
      cA[M * CSTR + HDIM + N] = (_Float16)(r * hid[M * OUTD + N]);
    }
    __syncthreads();   // B3: r*hid ready

    v8f hacc = {};
#pragma unroll
    for (int ks = 0; ks < 6; ++ks) {
      v16h af = ldfrag(arow + ks * 32 + half * 8);
      hacc = wmma_f16(af, ldfrag(hrow + ks * 32 + half * 8), hacc);
    }
    __syncthreads();   // B4: h-gemm reads done before hid rewrite

#pragma unroll
    for (int v = 0; v < 8; ++v) {
      int M = mt * 16 + v + 8 * half;
      float ht = hacc[v] + bhv;
      ht = ht > 0.f ? ht : 0.01f * ht;
      float hn = (1.f - zv[v]) * hid[M * OUTD + N] + zv[v] * ht;
      hid[M * OUTD + N] = hn;
      cA[M * CSTR + HDIM + N] = (_Float16)hn;
    }
    __syncthreads();
  }
  for (int i = tid; i < BGR * OUTD; i += 256) out[i] = hid[i];
}

// ---------------- host ----------------
static inline size_t alup(size_t x) { return (x + 255) & ~(size_t)255; }

extern "C" void kernel_launch(void* const* d_in, const int* in_sizes, int n_in,
                              void* d_out, int out_size, void* d_ws, size_t ws_size,
                              hipStream_t stream) {
  const float* x    = (const float*)d_in[0];
  const int*   nidx = (const int*)d_in[1];
  const int*   src  = (const int*)d_in[2];
  const int*   dst  = src + NEDGES;
  const float* emb  = (const float*)d_in[3];
  const float* Wl[3]  = { (const float*)d_in[4],  (const float*)d_in[8],  (const float*)d_in[12] };
  const float* asl[3] = { (const float*)d_in[5],  (const float*)d_in[9],  (const float*)d_in[13] };
  const float* adl[3] = { (const float*)d_in[6],  (const float*)d_in[10], (const float*)d_in[14] };
  const float* bl[3]  = { (const float*)d_in[7],  (const float*)d_in[11], (const float*)d_in[15] };
  const float* fcW = (const float*)d_in[16];
  const float* fcb = (const float*)d_in[17];
  const float* Wz = (const float*)d_in[18]; const float* bz = (const float*)d_in[19];
  const float* Wr = (const float*)d_in[20]; const float* br = (const float*)d_in[21];
  const float* Wh = (const float*)d_in[22]; const float* bh = (const float*)d_in[23];
  float* outp = (float*)d_out;

  char* ws = (char*)d_ws;
  size_t off = 0;
  _Float16* o_gru  = (_Float16*)(ws + off); off = alup(off + (size_t)EPER * BGR * HDIM * 2);
  _Float16* o_h0   = o_gru;  // aliased: h0f16 dead before gru buffer written
  _Float16* o_hf16 = (_Float16*)(ws + off); off = alup(off + (size_t)NNODES * HDIM * 2);
  float*    o_h    = (float*)(ws + off);    off = alup(off + (size_t)NNODES * HDIM * 4);
  float*    o_out  = (float*)(ws + off);    off = alup(off + (size_t)NNODES * HDIM * 4);
  float*    o_ss   = (float*)(ws + off);    off = alup(off + (size_t)NNODES * 4);
  float*    o_sd   = (float*)(ws + off);    off = alup(off + (size_t)NNODES * 4);
  unsigned* o_m    = (unsigned*)(ws + off); off = alup(off + (size_t)NNODES * 4);
  float*    o_sum  = (float*)(ws + off);    off = alup(off + (size_t)NNODES * 4);
  float*    o_ex   = (float*)(ws + off);    off = alup(off + (size_t)NEP * 4);
  _Float16* o_bt   = (_Float16*)(ws + off); off = alup(off + (size_t)128 * 256 * 2);
  _Float16* o_gwt  = (_Float16*)(ws + off); off = alup(off + (size_t)3 * 64 * 192 * 2);
  (void)ws_size; (void)in_sizes; (void)n_in; (void)out_size;

  // h0 = concat(x, emb[node_index]) f16 [NN][64]
  k_build_h0<<<(NNODES * 64) / 256, 256, 0, stream>>>(x, nidx, emb, o_h0);

  for (int layer = 0; layer < 3; ++layer) {
    const int K = (layer == 0) ? 64 : HDIM;
    const _Float16* Ain = (layer == 0) ? o_h0 : o_hf16;
    k_conv_bt<<<(K * HDIM + 255) / 256, 256, 0, stream>>>(Wl[layer], o_bt, K, HDIM);
    k_gemm_nodes<<<NNODES / 128, 256, 0, stream>>>(Ain, o_bt, o_h, K);
    k_scores<<<NNODES / 256, 256, 0, stream>>>(o_h, asl[layer], adl[layer], o_ss, o_sd);
    k_init_ms<<<NNODES / 256, 256, 0, stream>>>(o_m, o_sum);
    k_zero_f32<<<(NNODES * HDIM) / 256, 256, 0, stream>>>(o_out, NNODES * HDIM);
    k_edge_max<<<NEP / 256, 256, 0, stream>>>(o_ss, o_sd, src, dst, o_m);
    k_edge_expsum<<<NEP / 256, 256, 0, stream>>>(o_ss, o_sd, src, dst, o_m, o_ex, o_sum);
    k_aggregate<<<NEP / 2, 256, 0, stream>>>(o_h, src, dst, o_ex, o_sum, o_out);
    k_finalize<<<(NNODES * HDIM) / 256, 256, 0, stream>>>(o_out, bl[layer], o_hf16);
  }

  // edge FC -> gru input (time-major f16)
  k_conv_bt<<<(256 * HDIM + 255) / 256, 256, 0, stream>>>(fcW, o_bt, 256, HDIM);
  k_fc_gemm<<<NEDGES / 128, 256, 0, stream>>>(o_hf16, o_bt, src, dst, fcb, o_gru);

  // GRU weights -> transposed f16
  k_conv_bt<<<(192 * 64 + 255) / 256, 256, 0, stream>>>(Wz, o_gwt, 192, 64);
  k_conv_bt<<<(192 * 64 + 255) / 256, 256, 0, stream>>>(Wr, o_gwt + 64 * 192, 192, 64);
  k_conv_bt<<<(192 * 64 + 255) / 256, 256, 0, stream>>>(Wh, o_gwt + 2 * 64 * 192, 192, 64);

  k_gru<<<1, 256, 0, stream>>>(o_gru, o_gwt, bz, br, bh, outp);
}
